// RNNlayer_62569083568453
// MI455X (gfx1250) — matrix operand order repaired and checked by reading, not compiled, not run on hardware
//
#include <hip/hip_runtime.h>

// CDNA5 / gfx1250 wave32 WMMA types
typedef __attribute__((ext_vector_type(16))) _Float16 v16h;
typedef __attribute__((ext_vector_type(8)))  _Float16 v8h;
typedef __attribute__((ext_vector_type(8)))  float    v8f;
typedef __attribute__((ext_vector_type(4)))  int      v4i;

// Address-space-qualified pointer types for the async-to-LDS builtin:
// param0: v4i* in global (AS1), param1: v4i* in LDS (AS3).
typedef __attribute__((address_space(1))) v4i* gp_v4i;
typedef __attribute__((address_space(3))) v4i* lp_v4i;

#define BB 64
#define TT 2048
#define II 64
#define HH 512
#define KLDS 8        // W_hidden K-tiles resident in LDS (of 16)

// ---------------------------------------------------------------------------
// Kernel 0: repack W_in [I,H] and W_hidden [H,H] (f32, row-major) into the
// per-lane f16 B-fragment order this file uses (contiguous K, matched by its A loads) for v_wmma_f32_16x16x32_f16:
//   B is 32x16 (KxN); lane L holds column n = L&15, K base kb = (L>=16)?16:0;
//   vector element j <-> K = kb + j.  pack[nt][kt][lane][j]: 32B per lane.
// ---------------------------------------------------------------------------
__global__ __launch_bounds__(256) void pack_weights(const float* __restrict__ Win,
                                                    const float* __restrict__ Wh,
                                                    _Float16* __restrict__ packI,
                                                    _Float16* __restrict__ packH) {
  int tid = blockIdx.x * blockDim.x + threadIdx.x;
  int stride = gridDim.x * blockDim.x;
  const int totalH = 32 * 16 * 32 * 16;           // nt * kt * lane * j
  for (int idx = tid; idx < totalH; idx += stride) {
    int j = idx & 15, lane = (idx >> 4) & 31, kt = (idx >> 9) & 15, nt = idx >> 13;
    int kb = (lane >> 4) * 16;
    int k = kt * 32 + kb + j;
    int n = nt * 16 + (lane & 15);
    packH[idx] = (_Float16)Wh[k * HH + n];
  }
  const int totalI = 32 * 2 * 32 * 16;            // K = I = 64 -> 2 k-tiles
  for (int idx = tid; idx < totalI; idx += stride) {
    int j = idx & 15, lane = (idx >> 4) & 31, kt = (idx >> 9) & 1, nt = idx >> 10;
    int kb = (lane >> 4) * 16;
    int k = kt * 32 + kb + j;
    int n = nt * 16 + (lane & 15);
    packI[idx] = (_Float16)Win[k * HH + n];
  }
}

// ---------------------------------------------------------------------------
// Kernel 1: serial scan. 4 workgroups (one per 16-row batch block), 1024
// threads = 32 waves; wave w owns N-tile nt = w.  h lives in REGISTERS as the
// C/D fragment (8 f32/lane).  tanh(h) goes to double-buffered LDS f16 A-tiles
// (ping-pong on t&1 -> single barrier per step).  W_hidden B-fragments:
// K-tiles 0..7 in LDS (filled once via async-to-LDS), K-tiles 8..15 and both
// W_in tiles streamed from the L2-resident packed global buffers (W_in loads
// made LICM-opaque so they stay in-loop L2 reads instead of spilled regs).
// ---------------------------------------------------------------------------
__global__ __launch_bounds__(1024, 1) void rnn_scan(const float* __restrict__ x,
                                                    const float* __restrict__ sigma,
                                                    const float* __restrict__ h0,
                                                    const _Float16* __restrict__ packI,
                                                    const _Float16* __restrict__ packH,
                                                    float* __restrict__ out) {
  __shared__ _Float16 ldsBH[KLDS * 32 * 32 * 16];  // 256 KB: [nt*KLDS+kt][lane][j]
  __shared__ _Float16 aTile[2][16][HH];            //  32 KB: tanh(h) f16, ping-pong
  __shared__ _Float16 xTile[2][16][II];            //   4 KB: x[:,t,:] f16, ping-pong

  const int tid   = threadIdx.x;       // 0..1023
  const int wg    = blockIdx.x;        // 0..3 -> batch rows [16*wg, 16*wg+16)
  const int lane  = tid & 31;
  const int nt    = tid >> 5;          // wave id == N-tile, 0..31
  const int m     = lane & 15;         // A-fragment row
  const int kb0   = (lane >> 4) * 16;  // A-fragment K sub-offset per lane half
  const int ncol  = nt * 16 + (lane & 15);
  const int mbase = (lane >> 4) * 8;   // C/D fragment row base

  // ---- one-time fill of LDS with K-tiles 0..KLDS-1 of packed W_hidden.
  // 16384 x 16B chunks; 1024 threads x 16 iters. LDS frag (nt,kt) at
  // (nt*KLDS+kt)*1KB; global frag at (nt*16+kt)*1KB.
  for (int i = 0; i < (KLDS * 32 * 32 * 16 * 2) / 16 / 1024; ++i) {
    int f = tid + i * 1024;                       // b128 chunk index
    int fragIdx = f >> 6;                         // 64 chunks per 1KB fragment
    int within  = (f & 63) << 4;                  // byte offset inside fragment
    int fnt = fragIdx / KLDS, fkt = fragIdx % KLDS;
    char*       src = (char*)packH + ((fnt * 16 + fkt) << 10) + within;
    char*       dst = (char*)ldsBH + (fragIdx << 10) + within;
#if __has_builtin(__builtin_amdgcn_global_load_async_to_lds_b128)
    __builtin_amdgcn_global_load_async_to_lds_b128(
        (gp_v4i)(v4i*)(void*)src, (lp_v4i)(v4i*)(void*)dst, 0, 0);
#else
    *(v8h*)dst = *(const v8h*)src;
#endif
  }
#if __has_builtin(__builtin_amdgcn_global_load_async_to_lds_b128)
#if __has_builtin(__builtin_amdgcn_s_wait_asynccnt)
  __builtin_amdgcn_s_wait_asynccnt(0);
#else
  asm volatile("s_wait_asynccnt 0" ::: "memory");
#endif
#endif

  // ---- h fragment in registers (C/D layout): hfrag[r] = h[mbase+r][ncol]
  v8f hfrag;
#pragma unroll
  for (int r = 0; r < 8; ++r)
    hfrag[r] = h0[(wg * 16 + mbase + r) * HH + ncol];

  __syncthreads();   // LDS weight fill + h init complete

  const float c1 = 0.9f;   // 1 - DT/TAU
  const float c2 = 0.1f;   // DT/TAU

  for (int t = 0; t < TT; ++t) {
    const int buf = t & 1;
    // -------- phase 1: each wave writes tanh of its own h tile (f16) ------
#pragma unroll
    for (int r = 0; r < 8; ++r)
      aTile[buf][mbase + r][ncol] = (_Float16)tanhf(hfrag[r]);
    {
      int rr = tid >> 6, cc = tid & 63;  // 16*64 == 1024: one element/thread
      xTile[buf][rr][cc] = (_Float16)x[((wg * 16 + rr) * TT + t) * II + cc];
    }
    {  // prefetch next step's 32KB sigma tile (full coverage across threads)
      int t1 = (t + 1 < TT) ? (t + 1) : t;
      long offp = ((long)(wg * 16 + (tid >> 6)) * TT + t1) * HH + ((tid & 63) << 3);
      __builtin_prefetch(sigma + offp, 0, 1);
    }
    __syncthreads();   // sole barrier per step (ping-pong makes tail barrier
                       // unnecessary: buffer buf is next overwritten only at
                       // step t+2, gated by the step-t+1 barrier)

    // -------- phase 2: acc = x_t @ W_in + tanh(h) @ W_hidden --------------
    v8f acc = {};
#pragma unroll
    for (int kt = 0; kt < 2; ++kt) {             // W_in tiles (L2-streamed)
      v8h lo = *reinterpret_cast<const v8h*>(&xTile[buf][m][kt * 32 + kb0]);
      v8h hi = *reinterpret_cast<const v8h*>(&xTile[buf][m][kt * 32 + 8 + kb0]);
      v16h a = __builtin_shufflevector(lo, hi, 0, 1, 2, 3, 4, 5, 6, 7,
                                               8, 9, 10, 11, 12, 13, 14, 15);
      // LICM-opaque pointer: keep this a per-iteration L2 load, never a
      // hoisted (and then spilled) register-resident value.
      const _Float16* pI = packI + (((nt * 2 + kt) * 32 + lane) << 4);
      asm volatile("" : "+v"(pI));
      v16h b = *reinterpret_cast<const v16h*>(pI);
      acc = __builtin_amdgcn_wmma_f32_16x16x32_f16(false, a, false, b,
                                                   (short)0, acc, false, false);
    }
#pragma unroll
    for (int kt = 0; kt < KLDS; ++kt) {          // LDS-resident W_hidden tiles
      v8h lo = *reinterpret_cast<const v8h*>(&aTile[buf][m][kt * 32 + kb0]);
      v8h hi = *reinterpret_cast<const v8h*>(&aTile[buf][m][kt * 32 + 8 + kb0]);
      v16h a = __builtin_shufflevector(lo, hi, 0, 1, 2, 3, 4, 5, 6, 7,
                                               8, 9, 10, 11, 12, 13, 14, 15);
      v16h b = *reinterpret_cast<const v16h*>(ldsBH + (((nt * KLDS + kt) * 32 + lane) << 4));
      acc = __builtin_amdgcn_wmma_f32_16x16x32_f16(false, a, false, b,
                                                   (short)0, acc, false, false);
    }
#pragma unroll
    for (int kt = KLDS; kt < 16; ++kt) {         // L2-streamed W_hidden tiles
      v8h lo = *reinterpret_cast<const v8h*>(&aTile[buf][m][kt * 32 + kb0]);
      v8h hi = *reinterpret_cast<const v8h*>(&aTile[buf][m][kt * 32 + 8 + kb0]);
      v16h a = __builtin_shufflevector(lo, hi, 0, 1, 2, 3, 4, 5, 6, 7,
                                               8, 9, 10, 11, 12, 13, 14, 15);
      v16h b = *reinterpret_cast<const v16h*>(packH + (((nt * 16 + kt) * 32 + lane) << 4));
      acc = __builtin_amdgcn_wmma_f32_16x16x32_f16(false, a, false, b,
                                                   (short)0, acc, false, false);
    }

    // -------- phase 3: h = c1*h + c2*acc + sigma; store out (registers) ---
#pragma unroll
    for (int r = 0; r < 8; ++r) {
      long off = ((long)(wg * 16 + mbase + r) * TT + t) * HH + ncol;
      float hnew = c1 * hfrag[r] + c2 * acc[r] + sigma[off];
      out[off] = hnew;
      hfrag[r] = hnew;
    }
  }

  // ---- h_f appended after out in d_out
  float* hf = out + (long)BB * TT * HH;
#pragma unroll
  for (int r = 0; r < 8; ++r)
    hf[(wg * 16 + mbase + r) * HH + ncol] = hfrag[r];
}

// ---------------------------------------------------------------------------
// Inputs: 0=x [B,T,I], 1=W_in [I,H], 2=W_hidden [H,H], 3=sigma_all [B,T,H],
// 4=h_0 [B,H] (all f32).  Output: out [B,T,H] then h_f [B,H], concatenated.
// Workspace: packI (64 KB f16) + packH (512 KB f16).
// ---------------------------------------------------------------------------
extern "C" void kernel_launch(void* const* d_in, const int* in_sizes, int n_in,
                              void* d_out, int out_size, void* d_ws, size_t ws_size,
                              hipStream_t stream) {
  const float* x     = (const float*)d_in[0];
  const float* Win   = (const float*)d_in[1];
  const float* Wh    = (const float*)d_in[2];
  const float* sigma = (const float*)d_in[3];
  const float* h0    = (const float*)d_in[4];
  float* out = (float*)d_out;

  _Float16* packI = (_Float16*)d_ws;
  _Float16* packH = packI + 32 * 2 * 32 * 16;     // after 64 KB of W_in pack

  pack_weights<<<256, 256, 0, stream>>>(Win, Wh, packI, packH);
  rnn_scan<<<4, 1024, 0, stream>>>(x, sigma, h0, packI, packH, out);
}
